// MoEBlock_85392539779143
// MI455X (gfx1250) — compile-verified, matrix operand order checked by reading, not run on hardware
//
#include <hip/hip_runtime.h>
#include <hip/hip_bf16.h>

typedef __attribute__((ext_vector_type(16))) _Float16 v16h;
typedef __attribute__((ext_vector_type(8)))  float    v8f;

constexpr int kB  = 4096;   // tokens
constexpr int kD  = 1024;   // model dim
constexpr int kH  = 4096;   // hidden dim
constexpr int kE  = 8;      // experts
constexpr int kMB = 64;     // tokens per fast-FFN workgroup (4 WMMA M-tiles)
constexpr int kHC = 256;    // H-chunk resident in LDS
constexpr int kMaxATiles = 544;  // >= sum_e 4*ceil(cnt_e/64)

constexpr int kW1Frags = kE * (kH / 16) * (kD / 32);   // 65536
constexpr int kW2Frags = kE * (kD / 16) * (kH / 32);   // 65536

// K-pair base for 16-bit WMMA A/B fragments (ISA 7.12.2):
__device__ __forceinline__ int kpair_base(int v, int grp) {
    return (v < 4 ? 2 * v : 16 + 2 * (v - 4)) + 8 * grp;
}
// element (m, k) of a 16x32 fragment -> flat index, lane-major
// (lane*16 + elem) so each lane's 16 halves are one contiguous 32B block.
__device__ __forceinline__ int frag_idx(int m, int k) {
    int grp  = (k >> 3) & 1;
    int v    = (k & 16) ? 4 + ((k & 7) >> 1) : ((k & 7) >> 1);
    return (m + 16 * grp) * 16 + 2 * v + (k & 1);
}

__global__ void zero_kernel(float* __restrict__ out, int n, int* __restrict__ counts) {
    int gid = blockIdx.x * blockDim.x + threadIdx.x;
    if (gid < n) out[gid] = 0.0f;
    if (gid < kE) counts[gid] = 0;
}

__global__ void router_kernel(const float* __restrict__ x,
                              const float* __restrict__ Wr,
                              const float* __restrict__ br,
                              int* __restrict__ counts,
                              int* __restrict__ tlist,
                              float* __restrict__ gates) {
    int b = blockIdx.x * blockDim.x + threadIdx.x;
    if (b >= kB) return;

    float acc[kE];
#pragma unroll
    for (int j = 0; j < kE; ++j) acc[j] = br[j];
    const float* xr = x + (size_t)b * kD;
    for (int d = 0; d < kD; ++d) {
        float xv = xr[d];
#pragma unroll
        for (int j = 0; j < kE; ++j) acc[j] = fmaf(xv, Wr[d * kE + j], acc[j]);
    }
    int i0 = 0;
#pragma unroll
    for (int j = 1; j < kE; ++j) if (acc[j] > acc[i0]) i0 = j;
    int i1 = (i0 == 0) ? 1 : 0;
#pragma unroll
    for (int j = 0; j < kE; ++j) if (j != i0 && acc[j] > acc[i1]) i1 = j;

    float e1 = __expf(acc[i1] - acc[i0]);
    float w0 = 1.0f / (1.0f + e1);
    float w1 = e1 * w0;

    int s0 = atomicAdd(&counts[i0], 1);
    tlist[i0 * kB + s0] = b; gates[i0 * kB + s0] = w0;
    int s1 = atomicAdd(&counts[i1], 1);
    tlist[i1 * kB + s1] = b; gates[i1 * kB + s1] = w1;
}

__global__ void prefix_kernel(const int* __restrict__ counts, int* __restrict__ tileOff) {
    if (blockIdx.x == 0 && threadIdx.x == 0) {
        int acc = 0;
        for (int e = 0; e < kE; ++e) {
            tileOff[e] = acc;
            acc += 4 * ((counts[e] + 63) / 64);   // atiles per expert (padded to 64 rows)
        }
    }
}

__global__ void __launch_bounds__(256)
gather_x_kernel(const float* __restrict__ x,
                const int* __restrict__ counts, const int* __restrict__ tlist,
                const int* __restrict__ tileOff, _Float16* __restrict__ xg) {
    const int e  = blockIdx.y;
    const int at = blockIdx.x;
    const int cnt = counts[e];
    const int atiles = 4 * ((cnt + 63) / 64);
    if (at >= atiles) return;

    __shared__ int tokA[16];
    const int tid = threadIdx.x;
    if (tid < 16) {
        int slot = at * 16 + tid;
        tokA[tid] = (slot < cnt) ? tlist[e * kB + slot] : 0;
    }
    __syncthreads();

    const size_t base = (size_t)(tileOff[e] + at) * 32 * 512;
    for (int i = tid; i < 16 * kD; i += 256) {
        int r = i / kD, c = i % kD;
        xg[base + (size_t)(c >> 5) * 512 + frag_idx(r, c & 31)] =
            (_Float16)x[(size_t)tokA[r] * kD + c];
    }
}

__global__ void __launch_bounds__(256)
convert_w_kernel(const float* __restrict__ W1, const float* __restrict__ W2,
                 unsigned* __restrict__ w1h, unsigned* __restrict__ w2h) {
    int f = blockIdx.x, t = threadIdx.x;
    int lane = t >> 3, v = t & 7;
    int grp = lane >> 4, nsub = lane & 15;
    int k0 = kpair_base(v, grp);

    const float* src; unsigned* dst; int ntile, ktile, N, fl;
    if (f < kW1Frags) {
        fl = f;
        int e = fl >> 13, r = fl & 8191;
        ntile = r >> 5; ktile = r & 31; N = kH;
        src = W1 + (size_t)e * kD * kH; dst = w1h;
    } else {
        fl = f - kW1Frags;
        int e = fl >> 13, r = fl & 8191;
        ntile = r >> 7; ktile = r & 127; N = kD;
        src = W2 + (size_t)e * kH * kD; dst = w2h;
    }
    int k = ktile * 32 + k0;
    int n = ntile * 16 + nsub;
    union { _Float16 h[2]; unsigned u; } p;
    p.h[0] = (_Float16)src[(size_t)k * N + n];
    p.h[1] = (_Float16)src[(size_t)(k + 1) * N + n];
    dst[(size_t)fl * 256 + t] = p.u;
}

// ------------- fast FFN: M=64, 16 waves, fragment-major f16 operands -------------
__global__ void __launch_bounds__(512)
moe_ffn_fast_kernel(const v16h* __restrict__ xg,
                    const v16h* __restrict__ w1f, const float* __restrict__ b1,
                    const v16h* __restrict__ w2f, const float* __restrict__ b2,
                    const int*   __restrict__ counts,
                    const int*   __restrict__ tlist,
                    const float* __restrict__ gates,
                    const int*   __restrict__ tileOff,
                    float* __restrict__ out) {
    __shared__ __align__(32) _Float16 HsFrag[(kHC / 32) * 4 * 512];  // 32 KB
    __shared__ int   tokS[kMB];
    __shared__ float gateS[kMB];

    const int e = blockIdx.y;
    const int cnt = counts[e];
    if (blockIdx.x * kMB >= cnt) return;

    const int tid = threadIdx.x;
    if (tid < kMB) {
        int slot = blockIdx.x * kMB + tid;
        if (slot < cnt) { tokS[tid] = tlist[e * kB + slot]; gateS[tid] = gates[e * kB + slot]; }
        else            { tokS[tid] = 0;                    gateS[tid] = 0.0f; }
    }
    __syncthreads();

    const int wave = tid >> 5;       // 0..15
    const int lane = tid & 31;
    const int grp  = lane >> 4;
    const int nsub = lane & 15;

    const float* b1e = b1 + (size_t)e * kH;
    const float* b2e = b2 + (size_t)e * kD;

    const size_t abase = (size_t)(tileOff[e] + blockIdx.x * 4);   // 4 atiles / block

    v8f acc2[16] = {};   // [m=4][t=4]

    for (int hc = 0; hc < kH; hc += kHC) {
        // ---- Phase 1: h-chunk = relu(X @ W1[:, hc:hc+kHC] + b1) ----
        v8f acc1[4] = {};  // [m=4], one ntile per wave
        for (int kb = 0; kb < kD; kb += 32) {
            int kt = kb >> 5;
            v16h a0 = xg[((abase + 0) * 32 + kt) * 32 + lane];
            v16h a1 = xg[((abase + 1) * 32 + kt) * 32 + lane];
            v16h a2 = xg[((abase + 2) * 32 + kt) * 32 + lane];
            v16h a3 = xg[((abase + 3) * 32 + kt) * 32 + lane];
            int ntile = (hc >> 4) + wave;
            v16h bf = w1f[((size_t)(e * 256 + ntile) * 32 + kt) * 32 + lane];
            acc1[0] = __builtin_amdgcn_wmma_f32_16x16x32_f16(
                false, a0, false, bf, (short)0, acc1[0], false, false);
            acc1[1] = __builtin_amdgcn_wmma_f32_16x16x32_f16(
                false, a1, false, bf, (short)0, acc1[1], false, false);
            acc1[2] = __builtin_amdgcn_wmma_f32_16x16x32_f16(
                false, a2, false, bf, (short)0, acc1[2], false, false);
            acc1[3] = __builtin_amdgcn_wmma_f32_16x16x32_f16(
                false, a3, false, bf, (short)0, acc1[3], false, false);
        }
        {
            int lcol = wave * 16 + nsub;                 // 0..255 within chunk
            float bias = b1e[hc + lcol];
#pragma unroll
            for (int m = 0; m < 4; ++m)
#pragma unroll
                for (int r = 0; r < 8; ++r) {
                    float vv = acc1[m][r] + bias;
                    HsFrag[((lcol >> 5) * 4 + m) * 512 + frag_idx(r + 8 * grp, lcol & 31)] =
                        (_Float16)(vv > 0.0f ? vv : 0.0f);
                }
        }
        __syncthreads();

        // ---- Phase 2: acc2 += h-chunk @ W2[hc:hc+kHC, :] ----
        for (int kb = 0; kb < kHC; kb += 32) {
            int kt2 = kb >> 5;
            v16h a0 = *(const v16h*)(HsFrag + (kt2 * 4 + 0) * 512 + lane * 16);
            v16h a1 = *(const v16h*)(HsFrag + (kt2 * 4 + 1) * 512 + lane * 16);
            v16h a2 = *(const v16h*)(HsFrag + (kt2 * 4 + 2) * 512 + lane * 16);
            v16h a3 = *(const v16h*)(HsFrag + (kt2 * 4 + 3) * 512 + lane * 16);
            int ktg = (hc + kb) >> 5;
#pragma unroll
            for (int t = 0; t < 4; ++t) {
                v16h bf = w2f[((size_t)(e * 64 + wave * 4 + t) * 128 + ktg) * 32 + lane];
                acc2[0 * 4 + t] = __builtin_amdgcn_wmma_f32_16x16x32_f16(
                    false, a0, false, bf, (short)0, acc2[0 * 4 + t], false, false);
                acc2[1 * 4 + t] = __builtin_amdgcn_wmma_f32_16x16x32_f16(
                    false, a1, false, bf, (short)0, acc2[1 * 4 + t], false, false);
                acc2[2 * 4 + t] = __builtin_amdgcn_wmma_f32_16x16x32_f16(
                    false, a2, false, bf, (short)0, acc2[2 * 4 + t], false, false);
                acc2[3 * 4 + t] = __builtin_amdgcn_wmma_f32_16x16x32_f16(
                    false, a3, false, bf, (short)0, acc2[3 * 4 + t], false, false);
            }
        }
        __syncthreads();
    }

#pragma unroll
    for (int t = 0; t < 4; ++t) {
        int dcol = wave * 64 + t * 16 + nsub;
        float bias = b2e[dcol];
#pragma unroll
        for (int m = 0; m < 4; ++m)
#pragma unroll
            for (int r = 0; r < 8; ++r) {
                int row = m * 16 + r + 8 * grp;
                float vv = acc2[m * 4 + t][r] + bias;
                float s = 1.0f / (1.0f + __expf(-vv));
                atomicAdd(&out[(size_t)tokS[row] * kD + dcol], s * gateS[row]);
            }
    }
}

// ---------------- fallback FFN: direct fp32 weights (small workspace) ----------------
__global__ void __launch_bounds__(256)
moe_ffn_direct_kernel(const float* __restrict__ x,
                      const float* __restrict__ W1, const float* __restrict__ b1,
                      const float* __restrict__ W2, const float* __restrict__ b2,
                      const int*   __restrict__ counts,
                      const int*   __restrict__ tlist,
                      const float* __restrict__ gates,
                      float* __restrict__ out) {
    __shared__ _Float16 Xs[16][kD];
    __shared__ _Float16 Hs[16][512];
    __shared__ int      tokS[16];
    __shared__ float    gateS[16];

    const int e = blockIdx.y;
    const int cnt = counts[e];
    if (blockIdx.x * 16 >= cnt) return;

    const int tid = threadIdx.x;
    if (tid < 16) {
        int slot = blockIdx.x * 16 + tid;
        if (slot < cnt) { tokS[tid] = tlist[e * kB + slot]; gateS[tid] = gates[e * kB + slot]; }
        else            { tokS[tid] = 0;                    gateS[tid] = 0.0f; }
    }
    __syncthreads();
    for (int i = tid; i < 16 * kD; i += blockDim.x) {
        int r = i / kD, c = i % kD;
        Xs[r][c] = (_Float16)x[(size_t)tokS[r] * kD + c];
    }
    __syncthreads();

    const int wave = tid >> 5;
    const int lane = tid & 31;
    const int grp  = lane >> 4;
    const int nsub = lane & 15;

    const float* W1e = W1 + (size_t)e * kD * kH;
    const float* W2e = W2 + (size_t)e * kH * kD;
    const float* b1e = b1 + (size_t)e * kH;
    const float* b2e = b2 + (size_t)e * kD;

    v8f acc2[8] = {};
    for (int hc = 0; hc < kH; hc += 512) {
        v8f acc1[4] = {};
        for (int kb = 0; kb < kD; kb += 32) {
            v16h a;
#pragma unroll
            for (int v = 0; v < 8; ++v) {
                const _Float16* p = &Xs[nsub][kb + kpair_base(v, grp)];
                a[2 * v] = p[0]; a[2 * v + 1] = p[1];
            }
#pragma unroll
            for (int t = 0; t < 4; ++t) {
                int ncol = hc + wave * 64 + t * 16 + nsub;
                v16h bf;
#pragma unroll
                for (int v = 0; v < 8; ++v) {
                    int k = kb + kpair_base(v, grp);
                    bf[2 * v]     = (_Float16)W1e[(size_t)k       * kH + ncol];
                    bf[2 * v + 1] = (_Float16)W1e[(size_t)(k + 1) * kH + ncol];
                }
                acc1[t] = __builtin_amdgcn_wmma_f32_16x16x32_f16(
                    false, a, false, bf, (short)0, acc1[t], false, false);
            }
        }
#pragma unroll
        for (int t = 0; t < 4; ++t) {
            int lcol = wave * 64 + t * 16 + nsub;
            float bias = b1e[hc + lcol];
#pragma unroll
            for (int r = 0; r < 8; ++r) {
                float vv = acc1[t][r] + bias;
                Hs[r + 8 * grp][lcol] = (_Float16)(vv > 0.0f ? vv : 0.0f);
            }
        }
        __syncthreads();
        for (int kb = 0; kb < 512; kb += 32) {
            v16h a;
#pragma unroll
            for (int v = 0; v < 8; ++v) {
                const _Float16* p = &Hs[nsub][kb + kpair_base(v, grp)];
                a[2 * v] = p[0]; a[2 * v + 1] = p[1];
            }
#pragma unroll
            for (int t = 0; t < 8; ++t) {
                int dcol = wave * 128 + t * 16 + nsub;
                v16h bf;
#pragma unroll
                for (int v = 0; v < 8; ++v) {
                    int k = hc + kb + kpair_base(v, grp);
                    bf[2 * v]     = (_Float16)W2e[(size_t)k       * kD + dcol];
                    bf[2 * v + 1] = (_Float16)W2e[(size_t)(k + 1) * kD + dcol];
                }
                acc2[t] = __builtin_amdgcn_wmma_f32_16x16x32_f16(
                    false, a, false, bf, (short)0, acc2[t], false, false);
            }
        }
        __syncthreads();
    }
#pragma unroll
    for (int t = 0; t < 8; ++t) {
        int dcol = wave * 128 + t * 16 + nsub;
        float bias = b2e[dcol];
#pragma unroll
        for (int r = 0; r < 8; ++r) {
            int row = r + 8 * grp;
            float vv = acc2[t][r] + bias;
            float s = 1.0f / (1.0f + __expf(-vv));
            atomicAdd(&out[(size_t)tokS[row] * kD + dcol], s * gateS[row]);
        }
    }
}

extern "C" void kernel_launch(void* const* d_in, const int* in_sizes, int n_in,
                              void* d_out, int out_size, void* d_ws, size_t ws_size,
                              hipStream_t stream) {
    const float* x  = (const float*)d_in[0];
    const float* Wr = (const float*)d_in[1];
    const float* br = (const float*)d_in[2];
    const float* W1 = (const float*)d_in[3];
    const float* b1 = (const float*)d_in[4];
    const float* W2 = (const float*)d_in[5];
    const float* b2 = (const float*)d_in[6];
    float* out = (float*)d_out;

    int*   counts  = (int*)d_ws;
    int*   tlist   = counts + kE;
    float* gates   = (float*)(tlist + kE * kB);
    int*   tileOff = (int*)(gates + kE * kB);

    const size_t wOff    = (size_t)512 * 1024;
    const size_t wBytes  = (size_t)kE * kD * kH * sizeof(_Float16);      // 64 MB each
    const size_t xgBytes = (size_t)kMaxATiles * 16 * kD * sizeof(_Float16);
    const size_t wNeed   = wOff + 2 * wBytes + xgBytes;                  // ~147 MB
    _Float16* w1h = (_Float16*)((char*)d_ws + wOff);
    _Float16* w2h = w1h + (size_t)kE * kD * kH;
    _Float16* xgh = w2h + (size_t)kE * kH * kD;

    int n = kB * kD;
    zero_kernel<<<(n + 255) / 256, 256, 0, stream>>>(out, n, counts);
    router_kernel<<<kB / 256, 256, 0, stream>>>(x, Wr, br, counts, tlist, gates);

    if (ws_size >= wNeed) {
        prefix_kernel<<<1, 1, 0, stream>>>(counts, tileOff);
        convert_w_kernel<<<kW1Frags + kW2Frags, 256, 0, stream>>>(
            W1, W2, (unsigned*)w1h, (unsigned*)w2h);
        dim3 ggrid(4 * (kB / kMB), kE);   // max atiles per expert = 256
        gather_x_kernel<<<ggrid, 256, 0, stream>>>(x, counts, tlist, tileOff, xgh);
        dim3 grid(kB / kMB, kE);          // expert-major: weight reuse through L2
        moe_ffn_fast_kernel<<<grid, 512, 0, stream>>>(
            (const v16h*)xgh, (const v16h*)w1h, b1, (const v16h*)w2h, b2,
            counts, tlist, gates, tileOff, out);
    } else {
        dim3 grid(kB / 16, kE);
        moe_ffn_direct_kernel<<<grid, 256, 0, stream>>>(
            x, W1, b1, W2, b2, counts, tlist, gates, out);
    }
}
